// MemoryAttentionLayer_15101105013433
// MI455X (gfx1250) — matrix operand order repaired and checked by reading, not compiled, not run on hardware
//
#include <hip/hip_runtime.h>
#include <hip/hip_bf16.h>

#define B_    4
#define T_    512
#define H_    768
#define NM_   128
#define ROWS_ 16384
#define VPR_  64
#define KD_   128
#define KTOP_ 32
#define SST   65          // padded LDS score stride (64 banks -> conflict free)
#define ROWS_PER_BLOCK 8

typedef __attribute__((ext_vector_type(16))) __bf16 v16bf;
typedef __attribute__((ext_vector_type(8)))  float  v8f;

struct B32x8 { uint4 lo, hi; };

static __device__ __forceinline__ v16bf make_v16bf(uint4 lo, uint4 hi) {
    B32x8 u{lo, hi};
    return __builtin_bit_cast(v16bf, u);
}

// ---------------------------------------------------------------------------
// K1: gather start/end encodings, project to queries (f32 + bf16 copies).
// One block per mention, 128 threads (one per key dim).
// ---------------------------------------------------------------------------
__global__ __launch_bounds__(128) void query_proj_kernel(
    const float* __restrict__ enc,      // [B,T,H]
    const int*   __restrict__ mbp,
    const int*   __restrict__ msp,
    const int*   __restrict__ mep,
    const float* __restrict__ qw,       // [2H, KD]
    const float* __restrict__ qb,       // [KD]
    float*       __restrict__ qf,       // [NM, KD]
    __bf16*      __restrict__ qbf)      // [NM, KD]
{
    __shared__ float e[2 * H_];
    const int q = blockIdx.x;
    const int t = threadIdx.x;
    const int b = mbp[q], s = msp[q], ep = mep[q];
    const float* se = enc + ((size_t)b * T_ + s)  * H_;
    const float* ee = enc + ((size_t)b * T_ + ep) * H_;
    for (int i = t; i < H_; i += 128) { e[i] = se[i]; e[H_ + i] = ee[i]; }
    __syncthreads();
    float acc = qb[t];
    #pragma unroll 8
    for (int i = 0; i < 2 * H_; ++i) acc += e[i] * qw[i * KD_ + t];
    qf[q * KD_ + t]  = acc;
    qbf[q * KD_ + t] = (__bf16)acc;
}

// ---------------------------------------------------------------------------
// K2: bulk scoring with v_wmma_f32_16x16x32_bf16, software-pipelined.
// Block = 256 threads = 8 waves; wave w owns mentions [16w, 16w+16).
// Double-buffered bf16 key tile in LDS; row r+1 is staged from HBM into
// registers (outstanding on LOADcnt) while row r's 16 WMMAs run from LDS,
// then converted/stored to the alternate buffer. Keys stream from HBM once.
// ---------------------------------------------------------------------------
__global__ __launch_bounds__(256) void score_rows_kernel(
    const float*  __restrict__ mem_keys,   // [ROWS*VPR*KD] f32
    const __bf16* __restrict__ qbf,        // [NM*KD]
    float*        __restrict__ row_scores, // [NM*ROWS]
    int*          __restrict__ within_idx) // [NM*ROWS]
{
    __shared__ __bf16 kls[2][VPR_ * KD_];    // 2 x 16 KB bf16 key tiles
    __shared__ float  sls[8][16 * SST];      // per-wave 16x64 score tile (padded)

    const int tid    = threadIdx.x;
    const int wave   = tid >> 5;
    const int lane   = tid & 31;
    const int laneHi = lane >> 4;            // 0: K low half, 1: K high half
    const int lane16 = lane & 15;
    const int m0     = wave * 16;

    // Preload this wave's A fragments (16 mentions x K=0..127, bf16) into VGPRs.
    // A layout (matches B): lanes 0-15 = M, K=[0..15];
    // lanes 16-31 K=[16..31].
    v16bf afr[4];
    {
        const __bf16* qrow = qbf + (m0 + lane16) * KD_;
        #pragma unroll
        for (int ks = 0; ks < 4; ++ks) {
            const int ko = ks * 32;
            uint4 c0 = *(const uint4*)(qrow + ko +     laneHi * 16);
            uint4 c1 = *(const uint4*)(qrow + ko + 8 + laneHi * 16);
            afr[ks] = make_v16bf(c0, c1);
        }
    }

    const int row0 = blockIdx.x * ROWS_PER_BLOCK;

    // Prologue: stage row0 (each thread: 8 x float4 = 32 floats) and fill buf 0.
    float4 stage[8];
    {
        const float4* src = (const float4*)(mem_keys + (size_t)row0 * (VPR_ * KD_));
        #pragma unroll
        for (int j = 0; j < 8; ++j) stage[j] = src[tid + j * 256];
        #pragma unroll
        for (int j = 0; j < 8; ++j) {
            const int i = (tid + j * 256) * 4;
            kls[0][i + 0] = (__bf16)stage[j].x;
            kls[0][i + 1] = (__bf16)stage[j].y;
            kls[0][i + 2] = (__bf16)stage[j].z;
            kls[0][i + 3] = (__bf16)stage[j].w;
        }
    }
    __syncthreads();

    for (int rr = 0; rr < ROWS_PER_BLOCK; ++rr) {
        const int row = row0 + rr;
        const int cur = rr & 1;
        const bool hasNext = (rr + 1 < ROWS_PER_BLOCK);

        // Issue next row's global loads now; they stay outstanding (LOADcnt)
        // while the WMMAs below execute out of LDS.
        if (hasNext) {
            const float4* srcn =
                (const float4*)(mem_keys + (size_t)(row + 1) * (VPR_ * KD_));
            #pragma unroll
            for (int j = 0; j < 8; ++j) stage[j] = srcn[tid + j * 256];
        }

        // 4 N-tiles of 16 slots; batch the 4 K-step B fragments into VGPRs,
        // then chain 4 WMMAs back-to-back (one dscnt wait per chain).
        #pragma unroll
        for (int n = 0; n < 4; ++n) {
            const int slot = n * 16 + lane16;
            // B in this code's contiguous k order (same as A): lanes 0-15 hold K=[ko..ko+15] for column N,
            // lanes 16-31 hold K=[ko+16..ko+31]; dword j packs (K=2j, K=2j+1).
            const __bf16* krow = &kls[cur][slot * KD_ + laneHi * 16];
            v16bf bfr[4];
            #pragma unroll
            for (int ks = 0; ks < 4; ++ks)
                bfr[ks] = *(const v16bf*)(krow + ks * 32);
            v8f c = {};
            #pragma unroll
            for (int ks = 0; ks < 4; ++ks)
                c = __builtin_amdgcn_wmma_f32_16x16x32_bf16(
                        false, afr[ks], false, bfr[ks], (short)0, c, false, false);
            // D layout: VGPR g -> M = g + 8*laneHi, N = lane16.
            const int N  = n * 16 + lane16;
            const int Mb = laneHi * 8;
            #pragma unroll
            for (int g = 0; g < 8; ++g)
                sls[wave][(Mb + g) * SST + N] = c[g];
        }
        __syncthreads();

        // Per-(mention,row) max + argmax over the 64 slots; lanes 0-15 active.
        if (lane < 16) {
            const float* srow = &sls[wave][lane * SST];
            float best = -3.402823466e38f; int bi = 0;
            #pragma unroll 8
            for (int s = 0; s < VPR_; ++s) {
                float v = srow[s];
                if (v > best) { best = v; bi = s; }
            }
            const size_t o = (size_t)(m0 + lane) * ROWS_ + row;
            row_scores[o] = best;
            within_idx[o] = bi;
        }
        __syncthreads();

        // Epilogue of the pipeline stage: convert staged row into the other
        // buffer (the loadcnt wait lands here, after the compute phase).
        if (hasNext) {
            __bf16* dst = kls[cur ^ 1];
            #pragma unroll
            for (int j = 0; j < 8; ++j) {
                const int i = (tid + j * 256) * 4;
                dst[i + 0] = (__bf16)stage[j].x;
                dst[i + 1] = (__bf16)stage[j].y;
                dst[i + 2] = (__bf16)stage[j].z;
                dst[i + 3] = (__bf16)stage[j].w;
            }
            __syncthreads();
        }
    }
}

// ---------------------------------------------------------------------------
// K3: exact top-32 rows per query (32 masked block-argmax passes, L2-hot).
// Tie-break: lowest row index (matches jax.lax.top_k / argmax ordering).
// ---------------------------------------------------------------------------
__global__ __launch_bounds__(256) void topk_kernel(
    const float* __restrict__ rs,        // [NM*ROWS]
    const int*   __restrict__ wi,        // [NM*ROWS]
    int*         __restrict__ top_ids)   // [NM*KTOP]
{
    __shared__ float rv[256];
    __shared__ int   ri[256];
    __shared__ int   sel[KTOP_];
    const int q = blockIdx.x;
    const int t = threadIdx.x;
    const float* sc = rs + (size_t)q * ROWS_;

    for (int k = 0; k < KTOP_; ++k) {
        float best = -3.402823466e38f; int bi = -1;
        for (int r = t; r < ROWS_; r += 256) {
            bool taken = false;
            for (int j = 0; j < k; ++j) if (sel[j] == r) { taken = true; break; }
            if (taken) continue;
            float v = sc[r];
            if (v > best || (v == best && (unsigned)r < (unsigned)bi)) { best = v; bi = r; }
        }
        rv[t] = best; ri[t] = bi;
        __syncthreads();
        for (int st = 128; st > 0; st >>= 1) {
            if (t < st) {
                float ov = rv[t + st]; int oi = ri[t + st];
                if (ov > rv[t] || (ov == rv[t] && (unsigned)oi < (unsigned)ri[t])) {
                    rv[t] = ov; ri[t] = oi;
                }
            }
            __syncthreads();
        }
        if (t == 0) {
            const int r = ri[0];
            sel[k] = r;
            top_ids[q * KTOP_ + k] = r * VPR_ + wi[(size_t)q * ROWS_ + r];
        }
        __syncthreads();
    }
}

// ---------------------------------------------------------------------------
// K4: exact f32 attention over 32 gathered keys + update projection,
// atomic scatter-add into the updated-encoding buffer.
// ---------------------------------------------------------------------------
__global__ __launch_bounds__(128) void attn_update_kernel(
    const float* __restrict__ qf,        // [NM*KD]
    const int*   __restrict__ top_ids,   // [NM*KTOP]
    const float* __restrict__ mem_keys,  // flat [ROWS*VPR, KD]
    const float* __restrict__ mmask,     // [NM]
    const float* __restrict__ uw,        // [KD, H]
    const float* __restrict__ ub,        // [H]
    const int*   __restrict__ mbp,
    const int*   __restrict__ msp,
    float*       __restrict__ tmp)       // [B*T, H] (encoded copy)
{
    __shared__ float qs[KD_];
    __shared__ float kk[KTOP_][KD_];
    __shared__ int   ids[KTOP_];
    __shared__ float part[128];
    __shared__ float attn[KTOP_];
    __shared__ float rr[KD_];
    const int q = blockIdx.x;
    const int t = threadIdx.x;

    qs[t] = qf[q * KD_ + t];
    if (t < KTOP_) ids[t] = top_ids[q * KTOP_ + t];
    __syncthreads();
    for (int i = t; i < KTOP_ * KD_; i += 128) {
        const int k = i >> 7, d = i & (KD_ - 1);
        kk[k][d] = mem_keys[(size_t)ids[k] * KD_ + d];
    }
    __syncthreads();

    // scores (full-precision, with-gradient queries): 4 threads per key
    {
        const int k = t >> 2, p = t & 3;
        float s = 0.f;
        for (int d = p * 32; d < p * 32 + 32; ++d) s += qs[d] * kk[k][d];
        part[t] = s;
    }
    __syncthreads();
    if (t == 0) {
        float sco[KTOP_], mx = -3.402823466e38f;
        for (int k = 0; k < KTOP_; ++k) {
            sco[k] = part[4*k] + part[4*k+1] + part[4*k+2] + part[4*k+3];
            mx = fmaxf(mx, sco[k]);
        }
        float sum = 0.f;
        for (int k = 0; k < KTOP_; ++k) { sco[k] = __expf(sco[k] - mx); sum += sco[k]; }
        const float scale = mmask[q] / sum;              // mask folds into weights
        for (int k = 0; k < KTOP_; ++k) attn[k] = sco[k] * scale;
    }
    __syncthreads();

    // retrieved[d] = sum_k attn[k] * key[k][d]   (values == keys)
    {
        float r = 0.f;
        #pragma unroll 8
        for (int k = 0; k < KTOP_; ++k) r += attn[k] * kk[k][t];
        rr[t] = r;
    }
    __syncthreads();

    // update = retrieved @ uw + ub; scatter-add at (batch, start)
    const size_t base = ((size_t)mbp[q] * T_ + msp[q]) * H_;
    for (int o = t; o < H_; o += 128) {
        float u = ub[o];
        #pragma unroll 8
        for (int d = 0; d < KD_; ++d) u += rr[d] * uw[d * H_ + o];
        atomicAdd(&tmp[base + o], u);
    }
}

// ---------------------------------------------------------------------------
// K5: LayerNorm per token row.
// ---------------------------------------------------------------------------
__global__ __launch_bounds__(256) void ln_kernel(
    const float* __restrict__ x, const float* __restrict__ g,
    const float* __restrict__ be, float* __restrict__ out)
{
    __shared__ float red[256];
    const int row = blockIdx.x, t = threadIdx.x;
    const float* xr = x + (size_t)row * H_;
    float s = 0.f;
    for (int i = t; i < H_; i += 256) s += xr[i];
    red[t] = s; __syncthreads();
    for (int st = 128; st > 0; st >>= 1) { if (t < st) red[t] += red[t + st]; __syncthreads(); }
    const float mu = red[0] / (float)H_;
    __syncthreads();
    float v = 0.f;
    for (int i = t; i < H_; i += 256) { float d = xr[i] - mu; v += d * d; }
    red[t] = v; __syncthreads();
    for (int st = 128; st > 0; st >>= 1) { if (t < st) red[t] += red[t + st]; __syncthreads(); }
    const float inv = rsqrtf(red[0] / (float)H_ + 1e-12f);
    for (int i = t; i < H_; i += 256)
        out[(size_t)row * H_ + i] = (xr[i] - mu) * inv * g[i] + be[i];
}

// ---------------------------------------------------------------------------
extern "C" void kernel_launch(void* const* d_in, const int* in_sizes, int n_in,
                              void* d_out, int out_size, void* d_ws, size_t ws_size,
                              hipStream_t stream) {
    const float* enc   = (const float*)d_in[0];
    const int*   mbp   = (const int*)  d_in[1];
    const int*   msp   = (const int*)  d_in[2];
    const int*   mep   = (const int*)  d_in[3];
    const float* mmask = (const float*)d_in[4];
    const float* mkeys = (const float*)d_in[5];
    // d_in[6] = memory_entity_ids: unused by the reference output
    const float* qw    = (const float*)d_in[7];
    const float* qb    = (const float*)d_in[8];
    const float* uw    = (const float*)d_in[9];
    const float* ub    = (const float*)d_in[10];
    const float* g     = (const float*)d_in[11];
    const float* be    = (const float*)d_in[12];

    char* ws = (char*)d_ws;
    float*  qf   = (float*) (ws);                                  //  64 KB
    __bf16* qbf  = (__bf16*)(ws + 65536);                          //  32 KB
    float*  rs   = (float*) (ws + 98304);                          //   8 MB
    int*    wi   = (int*)   (ws + 98304 + 8u * 1024 * 1024);       //   8 MB
    int*    tids = (int*)   (ws + 98304 + 16u * 1024 * 1024);      //  16 KB
    float*  tmp  = (float*) (ws + 98304 + 16u * 1024 * 1024 + 65536); // 6 MB

    // Fresh copy of encoded_input to receive the scatter-add (deterministic per call).
    hipMemcpyAsync(tmp, enc, (size_t)B_ * T_ * H_ * sizeof(float),
                   hipMemcpyDeviceToDevice, stream);

    query_proj_kernel<<<NM_, 128, 0, stream>>>(enc, mbp, msp, mep, qw, qb, qf, qbf);
    score_rows_kernel<<<ROWS_ / ROWS_PER_BLOCK, 256, 0, stream>>>(mkeys, qbf, rs, wi);
    topk_kernel<<<NM_, 256, 0, stream>>>(rs, wi, tids);
    attn_update_kernel<<<NM_, 128, 0, stream>>>(qf, tids, mkeys, mmask, uw, ub,
                                                mbp, msp, tmp);
    ln_kernel<<<B_ * T_, 256, 0, stream>>>(tmp, g, be, (float*)d_out);
}
